// Sampler_base_24455543783469
// MI455X (gfx1250) — matrix operand order repaired and checked by reading, not compiled, not run on hardware
//
#include <hip/hip_runtime.h>

typedef __attribute__((ext_vector_type(16))) _Float16 v16h;
typedef __attribute__((ext_vector_type(8)))  _Float16 v8h;
typedef __attribute__((ext_vector_type(8)))  float    v8f;

// ---------------------------------------------------------------------------
// Shared-weight MLP layer on an LDS tile: Y = relu(X @ W^T + b)
//   X: LDS, rows x K (K multiple of 32), leading dim ldx (halfs)
//   W: global fp16, Cout rows x K cols (zero-padded), row-major
//   Y: LDS, rows x Cout, leading dim ldy
// One wave computes one 16x16 tile per task; waves of the block split tasks.
// Fragment layouts follow CDNA5 ISA 7.12.2 (wave32).
// ---------------------------------------------------------------------------
__device__ __forceinline__ void mlp_layer(const _Float16* X, int ldx, int K,
                                          const _Float16* W, const float* bias,
                                          int Cout, _Float16* Y, int ldy, int rows)
{
    const int lane   = threadIdx.x & 31;
    const int wave   = threadIdx.x >> 5;
    const int nwaves = blockDim.x >> 5;
    const int lo     = lane & 15;
    const int hi     = lane >> 4;
    const int rtiles = rows >> 4;
    const int ctiles = Cout >> 4;
    for (int t = wave; t < rtiles * ctiles; t += nwaves) {
        const int m0 = (t / ctiles) << 4;
        const int n0 = (t % ctiles) << 4;
        const int mr = m0 + lo;
        const int nn = n0 + lo;
        v8f acc = {};
        for (int k0 = 0; k0 < K; k0 += 32) {
            // A fragment: row mr, halves 0..7 -> K = k0+8*hi+i, 8..15 -> K = k0+16+8*hi+i
            const v8h* ap = (const v8h*)(X + (size_t)mr * ldx + k0 + hi * 8);
            v8h a_lo = ap[0];
            v8h a_hi = ap[2];   // +16 halfs
            // B fragment: col nn, halves 0..7 -> K = k0+8*hi+i, 8..15 -> K = k0+16+8*hi+i
            const v8h* bp = (const v8h*)(W + (size_t)nn * K + k0 + hi * 8);
            v8h b_lo = bp[0];
            v8h b_hi = bp[2];   // +16 halfs
            v16h a, b;
#pragma unroll
            for (int i = 0; i < 8; ++i) {
                a[i] = a_lo[i]; a[8 + i] = a_hi[i];
                b[i] = b_lo[i]; b[8 + i] = b_hi[i];
            }
            acc = __builtin_amdgcn_wmma_f32_16x16x32_f16(
                false, a, false, b, (short)0, acc, false, false);
        }
        const float bv = bias[nn];
#pragma unroll
        for (int r = 0; r < 8; ++r) {
            float v = acc[r] + bv;
            v = v > 0.f ? v : 0.f;
            // D: VGPR r holds row m0 + 8*hi + r, col nn
            Y[(size_t)(m0 + (hi << 3) + r) * ldy + nn] = (_Float16)v;
        }
    }
}

// ---------------------------------------------------------------------------
// fp32 weights (Cout x K) -> fp16 zero-padded (Cout x Kp)
// ---------------------------------------------------------------------------
__global__ void convert_w16(const float* __restrict__ W, _Float16* __restrict__ W16,
                            int Cout, int K, int Kp)
{
    int t = blockIdx.x * blockDim.x + threadIdx.x;
    if (t >= Cout * Kp) return;
    int r = t / Kp, c = t - r * Kp;
    W16[t] = (c < K) ? (_Float16)W[(size_t)r * K + c] : (_Float16)0.f;
}

// ---------------------------------------------------------------------------
// Farthest point sampling (one block per batch) + gather of selected coords.
// Matches reference: first selected index is 0; argmax with first-max tiebreak.
// ---------------------------------------------------------------------------
__global__ void fps_gather(const float* __restrict__ xyz, int N, int npoint,
                           float* __restrict__ new_xyz)
{
    __shared__ float dist[2048];
    __shared__ float rv[256];
    __shared__ int   ri[256];
    __shared__ int   lastIdx;
    const int b   = blockIdx.x;
    const int tid = threadIdx.x;
    const float* base = xyz + (size_t)b * N * 3;
    for (int i = tid; i < N; i += blockDim.x) dist[i] = 1e10f;
    if (tid == 0) lastIdx = 0;
    __syncthreads();
    for (int s = 0; s < npoint; ++s) {
        const int li = lastIdx;
        if (tid < 3) new_xyz[((size_t)b * npoint + s) * 3 + tid] = base[li * 3 + tid];
        const float px = base[li * 3 + 0];
        const float py = base[li * 3 + 1];
        const float pz = base[li * 3 + 2];
        float bestv = -1.f; int besti = 0;
        for (int i = tid; i < N; i += blockDim.x) {
            float dx = base[i * 3 + 0] - px;
            float dy = base[i * 3 + 1] - py;
            float dz = base[i * 3 + 2] - pz;
            float d  = dx * dx + dy * dy + dz * dz;
            float nd = fminf(dist[i], d);
            dist[i] = nd;
            if (nd > bestv) { bestv = nd; besti = i; }
        }
        rv[tid] = bestv; ri[tid] = besti;
        __syncthreads();
        for (int off = 128; off > 0; off >>= 1) {
            if (tid < off) {
                if (rv[tid + off] > rv[tid] ||
                    (rv[tid + off] == rv[tid] && ri[tid + off] < ri[tid])) {
                    rv[tid] = rv[tid + off]; ri[tid] = ri[tid + off];
                }
            }
            __syncthreads();
        }
        if (tid == 0) lastIdx = ri[0];
        __syncthreads();
    }
}

// ---------------------------------------------------------------------------
// Ball query: first `nsample` indices (in index order) with d2 <= r2, padded
// with the first hit (or 0 if no hit) — matches reference semantics.
// ---------------------------------------------------------------------------
__global__ void ball_query(const float* __restrict__ xyz, const float* __restrict__ new_xyz,
                           int B, int N, int npoint, float r2, int* __restrict__ out)
{
    int t = blockIdx.x * blockDim.x + threadIdx.x;
    if (t >= B * npoint) return;
    int b = t / npoint;
    float cx = new_xyz[(size_t)t * 3 + 0];
    float cy = new_xyz[(size_t)t * 3 + 1];
    float cz = new_xyz[(size_t)t * 3 + 2];
    const float* base = xyz + (size_t)b * N * 3;
    int* o = out + (size_t)t * 32;
    int cnt = 0, first = -1;
    for (int i = 0; i < N && cnt < 32; ++i) {
        float dx = base[i * 3 + 0] - cx;
        float dy = base[i * 3 + 1] - cy;
        float dz = base[i * 3 + 2] - cz;
        if (dx * dx + dy * dy + dz * dz <= r2) {
            if (first < 0) first = i;
            o[cnt++] = i;
        }
    }
    if (first < 0) first = 0;
    for (; cnt < 32; ++cnt) o[cnt] = first;
}

// ---------------------------------------------------------------------------
// Fused set-abstraction: one workgroup = one (batch, center) group of 32
// samples. Build 32 x Kp input in LDS, run 3-layer WMMA MLP, maxpool 32->1.
// ---------------------------------------------------------------------------
__global__ void sa_fused(const float* __restrict__ xyz, const float* __restrict__ feats,
                         const float* __restrict__ new_xyz, const int* __restrict__ ballidx,
                         const _Float16* __restrict__ W1, const float* __restrict__ b1,
                         int Kp, int C1,
                         const _Float16* __restrict__ W2, const float* __restrict__ b2, int C2,
                         const _Float16* __restrict__ W3, const float* __restrict__ b3, int C3,
                         int N, int npoint, int Cf, float* __restrict__ out)
{
    __shared__ __align__(16) _Float16 bufA[32 * 288];
    __shared__ __align__(16) _Float16 bufB[32 * 512];
    __shared__ int sidx[32];

    const int blk = blockIdx.x;          // b * npoint + p
    const int b   = blk / npoint;
    const int tid = threadIdx.x;

    if (tid < 32) sidx[tid] = ballidx[(size_t)blk * 32 + tid];
    __syncthreads();

    const float cx = new_xyz[(size_t)blk * 3 + 0];
    const float cy = new_xyz[(size_t)blk * 3 + 1];
    const float cz = new_xyz[(size_t)blk * 3 + 2];

    const int total = 32 * Kp;
    for (int e = tid; e < total; e += 256) {
        int s = e / Kp, c = e - s * Kp;
        int id = sidx[s];
        float v = 0.f;
        if (c < 3) {
            v = xyz[((size_t)b * N + id) * 3 + c] - (c == 0 ? cx : (c == 1 ? cy : cz));
        } else if (c < 3 + Cf) {
            v = feats[((size_t)b * N + id) * Cf + (c - 3)];
        }
        bufA[s * 288 + c] = (_Float16)v;
    }
    __syncthreads();
    mlp_layer(bufA, 288, Kp, W1, b1, C1, bufB, 512, 32);
    __syncthreads();
    mlp_layer(bufB, 512, C1, W2, b2, C2, bufA, 288, 32);
    __syncthreads();
    mlp_layer(bufA, 288, C2, W3, b3, C3, bufB, 512, 32);
    __syncthreads();
    for (int c = tid; c < C3; c += 256) {
        float m = (float)bufB[c];
        for (int s = 1; s < 32; ++s) {
            float q = (float)bufB[s * 512 + c];
            m = q > m ? q : m;
        }
        out[(size_t)blk * C3 + c] = m;
    }
}

// ---------------------------------------------------------------------------
// 3-NN search: per unknown point, 3 smallest squared distances to known set,
// normalized inverse-distance weights (reference order: ascending distance).
// ---------------------------------------------------------------------------
__global__ void three_nn(const float* __restrict__ unk, const float* __restrict__ kn,
                         int B, int n, int m, int* __restrict__ oid, float* __restrict__ ow)
{
    int t = blockIdx.x * blockDim.x + threadIdx.x;
    if (t >= B * n) return;
    int b = t / n;
    float ux = unk[(size_t)t * 3 + 0];
    float uy = unk[(size_t)t * 3 + 1];
    float uz = unk[(size_t)t * 3 + 2];
    const float* base = kn + (size_t)b * m * 3;
    float d0 = 1e30f, d1 = 1e30f, d2 = 1e30f;
    int   i0 = 0, i1 = 0, i2 = 0;
    for (int i = 0; i < m; ++i) {
        float dx = base[i * 3 + 0] - ux;
        float dy = base[i * 3 + 1] - uy;
        float dz = base[i * 3 + 2] - uz;
        float d  = dx * dx + dy * dy + dz * dz;
        if (d < d0)      { d2 = d1; i2 = i1; d1 = d0; i1 = i0; d0 = d; i0 = i; }
        else if (d < d1) { d2 = d1; i2 = i1; d1 = d;  i1 = i; }
        else if (d < d2) { d2 = d;  i2 = i; }
    }
    float w0 = 1.f / (d0 + 1e-8f);
    float w1 = 1.f / (d1 + 1e-8f);
    float w2 = 1.f / (d2 + 1e-8f);
    float s  = w0 + w1 + w2;
    ow[(size_t)t * 3 + 0] = w0 / s;
    ow[(size_t)t * 3 + 1] = w1 / s;
    ow[(size_t)t * 3 + 2] = w2 / s;
    oid[(size_t)t * 3 + 0] = i0;
    oid[(size_t)t * 3 + 1] = i1;
    oid[(size_t)t * 3 + 2] = i2;
}

// ---------------------------------------------------------------------------
// Fused feature propagation: one workgroup = 16 unknown points. Build
// [interp(C1) | unk_feats(C2)] in LDS, run 2-layer WMMA MLP, write fp32.
// ---------------------------------------------------------------------------
__global__ void fp_fused(const float* __restrict__ kn_feats, int C1,
                         const float* __restrict__ unk_feats, int C2,
                         const int* __restrict__ nn_idx, const float* __restrict__ nn_w,
                         const _Float16* __restrict__ W1, const float* __restrict__ b1,
                         const _Float16* __restrict__ W2, const float* __restrict__ b2,
                         int n, int m, float* __restrict__ out)
{
    __shared__ __align__(16) _Float16 bufA[16 * 768];
    __shared__ __align__(16) _Float16 bufB[16 * 256];
    __shared__ int   si[16 * 3];
    __shared__ float sw[16 * 3];

    const int tiles = n >> 4;
    const int blk = blockIdx.x;
    const int b   = blk / tiles;
    const int p0  = (blk - b * tiles) << 4;
    const int tid = threadIdx.x;

    if (tid < 48) {
        si[tid] = nn_idx[((size_t)b * n + p0) * 3 + tid];
        sw[tid] = nn_w [((size_t)b * n + p0) * 3 + tid];
    }
    __syncthreads();

    const int K = C1 + C2;
    for (int e = tid; e < 16 * K; e += 256) {
        int s = e / K, c = e - s * K;
        float v;
        if (c < C1) {
            const float* kf = kn_feats + (size_t)b * m * C1;
            v = sw[s * 3 + 0] * kf[(size_t)si[s * 3 + 0] * C1 + c] +
                sw[s * 3 + 1] * kf[(size_t)si[s * 3 + 1] * C1 + c] +
                sw[s * 3 + 2] * kf[(size_t)si[s * 3 + 2] * C1 + c];
        } else {
            v = unk_feats[((size_t)b * n + p0 + s) * C2 + (c - C1)];
        }
        bufA[s * 768 + c] = (_Float16)v;
    }
    __syncthreads();
    mlp_layer(bufA, 768, K, W1, b1, 256, bufB, 256, 16);
    __syncthreads();
    mlp_layer(bufB, 256, 256, W2, b2, 256, bufA, 768, 16);
    __syncthreads();
    for (int e = tid; e < 16 * 256; e += 256) {
        int s = e >> 8, c = e & 255;
        out[((size_t)b * n + p0 + s) * 256 + c] = (float)bufA[s * 768 + c];
    }
}

// ---------------------------------------------------------------------------
// Regression head: out = hint + [hint(3) | f1(256)] @ reg_W^T + reg_b
// ---------------------------------------------------------------------------
__global__ void reg_kernel(const float* __restrict__ hint, const float* __restrict__ g1,
                           const float* __restrict__ regW, const float* __restrict__ regb,
                           float* __restrict__ out, int total)
{
    int t = blockIdx.x * blockDim.x + threadIdx.x;
    if (t >= total) return;
    float h0 = hint[(size_t)t * 3 + 0];
    float h1 = hint[(size_t)t * 3 + 1];
    float h2 = hint[(size_t)t * 3 + 2];
    const float* f = g1 + (size_t)t * 256;
#pragma unroll
    for (int k = 0; k < 3; ++k) {
        const float* wr = regW + (size_t)k * 259;
        float acc = regb[k] + h0 * wr[0] + h1 * wr[1] + h2 * wr[2];
        for (int c = 0; c < 256; ++c) acc += f[c] * wr[3 + c];
        float h = (k == 0 ? h0 : (k == 1 ? h1 : h2));
        out[(size_t)t * 3 + k] = h + acc;
    }
}

// ---------------------------------------------------------------------------
extern "C" void kernel_launch(void* const* d_in, const int* in_sizes, int n_in,
                              void* d_out, int out_size, void* d_ws, size_t ws_size,
                              hipStream_t stream)
{
    (void)in_sizes; (void)n_in; (void)out_size; (void)ws_size;
    const int B = 8, N0 = 8192, S = 2048;

    const float* pc   = (const float*)d_in[0];
    const float* hint = (const float*)d_in[1];
    const float* saW[3][3]; const float* saB[3][3];
    int di = 2;
    for (int l = 0; l < 3; ++l)
        for (int j = 0; j < 3; ++j) {
            saW[l][j] = (const float*)d_in[di++];
            saB[l][j] = (const float*)d_in[di++];
        }
    const float* fpW[2][2]; const float* fpB[2][2];
    for (int l = 0; l < 2; ++l)
        for (int j = 0; j < 2; ++j) {
            fpW[l][j] = (const float*)d_in[di++];
            fpB[l][j] = (const float*)d_in[di++];
        }
    const float* regW = (const float*)d_in[di++];
    const float* regb = (const float*)d_in[di++];

    // ---- workspace carve (bump allocator, 256B aligned) ----
    char* wsp = (char*)d_ws;
    auto carve = [&](size_t bytes) -> void* {
        void* p = (void*)wsp;
        wsp += (bytes + 255) & ~(size_t)255;
        return p;
    };
    float* new2 = (float*)carve((size_t)B * 512 * 3 * 4);
    float* new3 = (float*)carve((size_t)B * 128 * 3 * 4);
    int*   bidx = (int*)  carve((size_t)B * 2048 * 32 * 4);
    float* f1   = (float*)carve((size_t)B * 2048 * 128 * 4);
    float* f2   = (float*)carve((size_t)B * 512 * 256 * 4);
    float* f3   = (float*)carve((size_t)B * 128 * 512 * 4);
    float* g2   = (float*)carve((size_t)B * 512 * 256 * 4);
    float* g1   = (float*)carve((size_t)B * 2048 * 256 * 4);
    int*   nni  = (int*)  carve((size_t)B * 2048 * 3 * 4);
    float* nnw  = (float*)carve((size_t)B * 2048 * 3 * 4);

    static const int saK [3][3] = {{3, 64, 64},   {131, 128, 128}, {259, 256, 256}};
    static const int saKp[3][3] = {{32, 64, 64},  {160, 128, 128}, {288, 256, 256}};
    static const int saC [3][3] = {{64, 64, 128}, {128, 128, 256}, {256, 256, 512}};
    _Float16* saW16[3][3];
    for (int l = 0; l < 3; ++l)
        for (int j = 0; j < 3; ++j) {
            saW16[l][j] = (_Float16*)carve((size_t)saC[l][j] * saKp[l][j] * 2);
            int tot = saC[l][j] * saKp[l][j];
            convert_w16<<<(tot + 255) / 256, 256, 0, stream>>>(
                saW[l][j], saW16[l][j], saC[l][j], saK[l][j], saKp[l][j]);
        }
    static const int fpK[2][2] = {{384, 256}, {768, 256}};
    _Float16* fpW16[2][2];
    for (int l = 0; l < 2; ++l)
        for (int j = 0; j < 2; ++j) {
            fpW16[l][j] = (_Float16*)carve((size_t)256 * fpK[l][j] * 2);
            int tot = 256 * fpK[l][j];
            convert_w16<<<(tot + 255) / 256, 256, 0, stream>>>(
                fpW[l][j], fpW16[l][j], 256, fpK[l][j], fpK[l][j]);
        }

    // ---- SA level 1: centers = hint (2048), xyz = pointcloud (8192) ----
    ball_query<<<(B * 2048 + 255) / 256, 256, 0, stream>>>(
        pc, hint, B, N0, 2048, 0.2f * 0.2f, bidx);
    sa_fused<<<B * 2048, 256, 0, stream>>>(
        pc, nullptr, hint, bidx,
        saW16[0][0], saB[0][0], 32, 64,
        saW16[0][1], saB[0][1], 64,
        saW16[0][2], saB[0][2], 128,
        N0, 2048, 0, f1);

    // ---- FPS over hint -> 512 centers ----
    fps_gather<<<B, 256, 0, stream>>>(hint, S, 512, new2);

    // ---- SA level 2 ----
    ball_query<<<(B * 512 + 255) / 256, 256, 0, stream>>>(
        hint, new2, B, 2048, 512, 0.4f * 0.4f, bidx);
    sa_fused<<<B * 512, 256, 0, stream>>>(
        hint, f1, new2, bidx,
        saW16[1][0], saB[1][0], 160, 128,
        saW16[1][1], saB[1][1], 128,
        saW16[1][2], saB[1][2], 256,
        2048, 512, 128, f2);

    // ---- FPS over new2 -> 128 centers ----
    fps_gather<<<B, 256, 0, stream>>>(new2, 512, 128, new3);

    // ---- SA level 3 ----
    ball_query<<<(B * 128 + 255) / 256, 256, 0, stream>>>(
        new2, new3, B, 512, 128, 0.8f * 0.8f, bidx);
    sa_fused<<<B * 128, 256, 0, stream>>>(
        new2, f2, new3, bidx,
        saW16[2][0], saB[2][0], 288, 256,
        saW16[2][1], saB[2][1], 256,
        saW16[2][2], saB[2][2], 512,
        512, 128, 256, f3);

    // ---- FP (level 2 <- level 3): K = 512 + 256 = 768, fp_params[1] ----
    three_nn<<<(B * 512 + 255) / 256, 256, 0, stream>>>(new2, new3, B, 512, 128, nni, nnw);
    fp_fused<<<B * (512 / 16), 256, 0, stream>>>(
        f3, 512, f2, 256, nni, nnw,
        fpW16[1][0], fpB[1][0], fpW16[1][1], fpB[1][1],
        512, 128, g2);

    // ---- FP (level 1 <- level 2): K = 256 + 128 = 384, fp_params[0] ----
    three_nn<<<(B * 2048 + 255) / 256, 256, 0, stream>>>(hint, new2, B, 2048, 512, nni, nnw);
    fp_fused<<<B * (2048 / 16), 256, 0, stream>>>(
        g2, 256, f1, 128, nni, nnw,
        fpW16[0][0], fpB[0][0], fpW16[0][1], fpB[0][1],
        2048, 512, g1);

    // ---- regression head ----
    reg_kernel<<<(B * 2048 + 255) / 256, 256, 0, stream>>>(
        hint, g1, regW, regb, (float*)d_out, B * 2048);
}
